// SwinTransformerBlock_77730318123423
// MI455X (gfx1250) — matrix operand order repaired and checked by reading, not compiled, not run on hardware
//
#include <hip/hip_runtime.h>
#include <hip/hip_bf16.h>

// ---------------------------------------------------------------------------
// Swin block for MI455X (gfx1250, wave32). All GEMMs + attention run on
// v_wmma_f32_16x16x32_bf16 (f32 accumulate). Intermediates stored bf16 to
// halve HBM traffic (workload is bandwidth-bound: ~96 GFLOP vs ~1.8 GB).
// GEMM stage-in uses GLOBAL_LOAD_ASYNC_TO_LDS_B128 (ASYNCcnt) — the CDNA5
// direct memory->LDS path, avoiding the VGPR round trip.
// ---------------------------------------------------------------------------

#define BB   32
#define HH   112
#define WW   112
#define CC   96
#define NHD  3
#define WS7  7
#define NN   49          // tokens per window
#define HD   32          // head dim
#define HID  384
#define NWIN 8192        // B * 256 windows
#define MTOT 401408      // B*H*W rows

typedef __attribute__((ext_vector_type(16))) __bf16 v16bf;
typedef __attribute__((ext_vector_type(8)))  float  v8f;

union FragBF { unsigned int u[8]; v16bf v; };

static __device__ __forceinline__ unsigned short f2bf(float f) {
    unsigned int u = __float_as_uint(f);
    unsigned int r = u + 0x7FFFu + ((u >> 16) & 1u);   // round-to-nearest-even
    return (unsigned short)(r >> 16);
}

// A-fragment (16x32 bf16) K offset for VGPR j, lane-half h (CDNA5 wave32 layout)
static __device__ __forceinline__ int kAoff(int j, int h) {
    return ((j & 4) ? 16 : 0) + h * 8 + 2 * (j & 3);
}

// CDNA5 async memory->LDS copy (16B), tracked by ASYNCcnt.
// lds_off: byte offset in LDS (generic shared pointer truncated to 32 bits).
static __device__ __forceinline__ void async_ld_b128(unsigned lds_off,
                                                     const void* gaddr) {
    asm volatile("global_load_async_to_lds_b128 %0, %1, off"
                 :: "v"(lds_off), "v"((unsigned long long)(uintptr_t)gaddr)
                 : "memory");
}
static __device__ __forceinline__ void wait_async0() {
    asm volatile("s_wait_asynccnt 0x0" ::: "memory");
}
static __device__ __forceinline__ unsigned lds_addr(const void* p) {
    return (unsigned)(uintptr_t)p;   // generic->LDS: low 32 bits are the offset
}

// ---------------------------------------------------------------------------
// Weight f32 -> bf16 conversion (qkv_w | proj_w | fc1_w | fc2_w packed)
// ---------------------------------------------------------------------------
__global__ __launch_bounds__(256) void conv_weights(
        const float* __restrict__ qkvw, const float* __restrict__ projw,
        const float* __restrict__ fc1w, const float* __restrict__ fc2w,
        unsigned short* __restrict__ o) {
    int i = blockIdx.x * 256 + threadIdx.x;   // total 110592
    float v;
    if      (i < 27648) v = qkvw[i];
    else if (i < 36864) v = projw[i - 27648];
    else if (i < 73728) v = fc1w[i - 36864];
    else                v = fc2w[i - 73728];
    o[i] = f2bf(v);
}

// ---------------------------------------------------------------------------
// LN1 fused with cyclic shift + window partition. One wave per token row.
// ---------------------------------------------------------------------------
__global__ __launch_bounds__(256) void ln1_window(
        const float* __restrict__ x, const float* __restrict__ g,
        const float* __restrict__ b, unsigned short* __restrict__ xw) {
    int lane = threadIdx.x & 31;
    int r = blockIdx.x * 8 + (threadIdx.x >> 5);
    int win = r / NN, t = r - win * NN;
    int bb = win >> 8, wrem = win & 255;
    int wy = wrem >> 4, wx = wrem & 15;
    int ty = t / WS7, tx = t - ty * WS7;
    int hh = wy * WS7 + ty + 3; if (hh >= HH) hh -= HH;
    int ww = wx * WS7 + tx + 3; if (ww >= WW) ww -= WW;
    const float* src = x + ((size_t)bb * (HH * WW) + hh * WW + ww) * CC;
    float v0 = src[lane], v1 = src[lane + 32], v2 = src[lane + 64];
    float s1 = v0 + v1 + v2;
    float s2 = v0 * v0 + v1 * v1 + v2 * v2;
    #pragma unroll
    for (int off = 16; off >= 1; off >>= 1) {
        s1 += __shfl_xor(s1, off, 32);
        s2 += __shfl_xor(s2, off, 32);
    }
    float mean = s1 * (1.0f / 96.0f);
    float var  = s2 * (1.0f / 96.0f) - mean * mean;
    float rstd = rsqrtf(var + 1e-5f);
    unsigned short* dst = xw + (size_t)r * CC;
    dst[lane]      = f2bf((v0 - mean) * rstd * g[lane]      + b[lane]);
    dst[lane + 32] = f2bf((v1 - mean) * rstd * g[lane + 32] + b[lane + 32]);
    dst[lane + 64] = f2bf((v2 - mean) * rstd * g[lane + 64] + b[lane + 64]);
}

// LN2: identity mapping, reads x1 from d_out, writes bf16 h2
__global__ __launch_bounds__(256) void ln2_kernel(
        const float* __restrict__ x1, const float* __restrict__ g,
        const float* __restrict__ b, unsigned short* __restrict__ h2) {
    int lane = threadIdx.x & 31;
    int r = blockIdx.x * 8 + (threadIdx.x >> 5);
    const float* src = x1 + (size_t)r * CC;
    float v0 = src[lane], v1 = src[lane + 32], v2 = src[lane + 64];
    float s1 = v0 + v1 + v2;
    float s2 = v0 * v0 + v1 * v1 + v2 * v2;
    #pragma unroll
    for (int off = 16; off >= 1; off >>= 1) {
        s1 += __shfl_xor(s1, off, 32);
        s2 += __shfl_xor(s2, off, 32);
    }
    float mean = s1 * (1.0f / 96.0f);
    float var  = s2 * (1.0f / 96.0f) - mean * mean;
    float rstd = rsqrtf(var + 1e-5f);
    unsigned short* dst = h2 + (size_t)r * CC;
    dst[lane]      = f2bf((v0 - mean) * rstd * g[lane]      + b[lane]);
    dst[lane + 32] = f2bf((v1 - mean) * rstd * g[lane + 32] + b[lane + 32]);
    dst[lane + 64] = f2bf((v2 - mean) * rstd * g[lane + 64] + b[lane + 64]);
}

// ---------------------------------------------------------------------------
// Generic bf16 WMMA GEMM: D[m,n] = sum_k A[m,k] * W[n,k] (+ epilogue)
// Block = 256 thr / 8 waves; block tile 128(M) x 32(N); K step 32.
// Waves: 4 (M) x 2 (N); each wave computes 32x16 = 2 WMMAs per K step,
// sharing one B fragment. Stage-in via async global->LDS.
// ---------------------------------------------------------------------------
#define GM_BF16 0   // + bias -> bf16          (QKV)
#define GM_GELU 1   // + bias, GELU -> bf16    (FC1)
#define GM_PROJ 2   // + bias + window-reverse scatter + x residual -> f32 d_out
#define GM_FC2  3   // + bias, accumulate into f32 d_out (x1 + y)

template <int MODE>
__global__ __launch_bounds__(256) void gemm_bf16(
        const unsigned short* __restrict__ A,
        const unsigned short* __restrict__ W,
        const float* __restrict__ bias,
        unsigned short* __restrict__ outb,
        float* __restrict__ dout,
        const float* __restrict__ resx,
        int M, int N, int K) {
    __shared__ unsigned short As[128 * 40];
    __shared__ unsigned short Wsh[32 * 40];
    const int tid  = threadIdx.x;
    const int lane = tid & 31;
    const int wv   = tid >> 5;
    const int wm   = wv & 3, wn = wv >> 2;     // 4 x 2 waves
    const int nTilesN = N >> 5;
    const int m0 = (blockIdx.x / nTilesN) << 7;
    const int n0 = (blockIdx.x % nTilesN) << 5;
    const int half = lane >> 4, l15 = lane & 15;

    v8f acc0 = {0.f, 0.f, 0.f, 0.f, 0.f, 0.f, 0.f, 0.f};
    v8f acc1 = acc0;

    // stage-in coordinates
    const int arow = tid >> 1;                 // 0..127
    const int akc  = (tid & 1) << 4;           // 0 or 16 (elements)
    const int wrow = tid >> 2;                 // 0..63 (only tid<128 used)
    const int wkc  = (tid & 3) << 3;           // 0,8,16,24

    for (int k0 = 0; k0 < K; k0 += 32) {
        // A tile (128 x 32): 2 async b128 per thread
        {
            const unsigned short* src = A + (size_t)(m0 + arow) * K + k0 + akc;
            unsigned dst = lds_addr(&As[arow * 40 + akc]);
            async_ld_b128(dst,      src);
            async_ld_b128(dst + 16, src + 8);
        }
        // W tile (32 x 32): 1 async b128 per thread, first 128 threads
        if (tid < 128) {
            const unsigned short* src = W + (size_t)(n0 + wrow) * K + k0 + wkc;
            async_ld_b128(lds_addr(&Wsh[wrow * 40 + wkc]), src);
        }
        wait_async0();
        __syncthreads();

        FragBF a0, a1, b;
        #pragma unroll
        for (int j = 0; j < 8; ++j) {
            int ka = kAoff(j, half);
            a0.u[j] = *(const unsigned int*)(&As[(wm * 32 + l15) * 40 + ka]);
            a1.u[j] = *(const unsigned int*)(&As[(wm * 32 + 16 + l15) * 40 + ka]);
            b.u[j]  = *(const unsigned int*)(&Wsh[(wn * 16 + l15) * 40 + ka]);
        }
        acc0 = __builtin_amdgcn_wmma_f32_16x16x32_bf16(
                   false, a0.v, false, b.v, (short)0, acc0, false, false);
        acc1 = __builtin_amdgcn_wmma_f32_16x16x32_bf16(
                   false, a1.v, false, b.v, (short)0, acc1, false, false);
        __syncthreads();
    }

    const int n = n0 + wn * 16 + l15;
    const float bn = bias[n];
    #pragma unroll
    for (int sub = 0; sub < 2; ++sub) {
        const v8f& acc = sub ? acc1 : acc0;
        #pragma unroll
        for (int j = 0; j < 8; ++j) {
            int m = m0 + wm * 32 + sub * 16 + j + 8 * half;
            if (MODE == GM_BF16) {
                outb[(size_t)m * N + n] = f2bf(acc[j] + bn);
            } else if (MODE == GM_GELU) {
                float v = acc[j] + bn;
                v = 0.5f * v * (1.0f + erff(v * 0.70710678118654752f));
                outb[(size_t)m * N + n] = f2bf(v);
            } else if (MODE == GM_FC2) {
                size_t idx = (size_t)m * N + n;
                dout[idx] = dout[idx] + acc[j] + bn;     // final = x1 + y
            } else {  // GM_PROJ: window-reverse + unshift scatter, add shortcut
                int win = m / NN, t = m - win * NN;
                int bb = win >> 8, wrem = win & 255;
                int wy = wrem >> 4, wx = wrem & 15;
                int ty = t / WS7, tx = t - ty * WS7;
                int hh = wy * WS7 + ty + 3;  if (hh >= HH) hh -= HH;
                int wwp = wx * WS7 + tx + 3; if (wwp >= WW) wwp -= WW;
                size_t idx = ((size_t)bb * (HH * WW) + hh * WW + wwp) * CC + n;
                dout[idx] = acc[j] + bn + resx[idx];     // x1 = shortcut + proj
            }
        }
    }
}

// ---------------------------------------------------------------------------
// Windowed attention: 1 block (4 waves) per (window, head).
// S = scale*(Q Kt) + rel_bias + shift_mask ; softmax ; O = P V. 49 padded to 64.
// ---------------------------------------------------------------------------
__global__ __launch_bounds__(128) void attn_kernel(
        const unsigned short* __restrict__ qkv,   // (401408, 288) bf16
        const int*   __restrict__ rel_index,      // (49,49)
        const float* __restrict__ rbt,            // (169,3)
        const float* __restrict__ amask,          // (256,49,49)
        unsigned short* __restrict__ attn_out) {  // (401408, 96) bf16
    __shared__ unsigned short Ps[64 * 72];
    const int lane  = threadIdx.x & 31;
    const int mtile = threadIdx.x >> 5;           // wave id = M tile (16 rows)
    const int win   = blockIdx.x % NWIN;
    const int head  = blockIdx.x / NWIN;
    const int half = lane >> 4, l15 = lane & 15;
    const size_t qbase = (size_t)win * NN * 288 + head * HD;

    // Q fragment (A: 16x32 over head dim)
    FragBF qa;
    {
        int t = mtile * 16 + l15;
        const unsigned short* qr = qkv + qbase + (size_t)t * 288;
        #pragma unroll
        for (int j = 0; j < 8; ++j)
            qa.u[j] = *(const unsigned int*)(qr + kAoff(j, half));
    }

    // S = Q Kt : 4 N-tiles, K=32 -> one WMMA each
    v8f sacc[4];
    #pragma unroll
    for (int nt = 0; nt < 4; ++nt) {
        FragBF kb;
        int s = nt * 16 + l15;
        const unsigned short* kr = qkv + qbase + 96 + (size_t)s * 288;
        #pragma unroll
        for (int j = 0; j < 8; ++j)
            kb.u[j] = *(const unsigned int*)(kr + kAoff(j, half));
        v8f z = {0.f, 0.f, 0.f, 0.f, 0.f, 0.f, 0.f, 0.f};
        sacc[nt] = __builtin_amdgcn_wmma_f32_16x16x32_bf16(
                       false, qa.v, false, kb.v, (short)0, z, false, false);
    }

    // fused scale + bias + mask + row softmax (rows live in half-wave lanes)
    const float scale = 0.17677669529663687f;     // 32^-0.5
    const float* maskrow = amask + (size_t)(win & 255) * (NN * NN);
    #pragma unroll
    for (int j = 0; j < 8; ++j) {
        int tt = mtile * 16 + j + 8 * half;
        #pragma unroll
        for (int nt = 0; nt < 4; ++nt) {
            int s = nt * 16 + l15;
            float add;
            if (tt < NN && s < NN) {
                int ri = rel_index[tt * NN + s];
                add = rbt[ri * NHD + head] + maskrow[tt * NN + s];
            } else {
                add = -1e30f;                     // pad cols -> exp = 0
            }
            sacc[nt][j] = sacc[nt][j] * scale + add;
        }
        float mv = fmaxf(fmaxf(sacc[0][j], sacc[1][j]), fmaxf(sacc[2][j], sacc[3][j]));
        #pragma unroll
        for (int off = 8; off >= 1; off >>= 1) mv = fmaxf(mv, __shfl_xor(mv, off, 32));
        float sum = 0.f;
        #pragma unroll
        for (int nt = 0; nt < 4; ++nt) {
            float e = __expf(sacc[nt][j] - mv);
            sacc[nt][j] = e;
            sum += e;
        }
        #pragma unroll
        for (int off = 8; off >= 1; off >>= 1) sum += __shfl_xor(sum, off, 32);
        float r = 1.0f / sum;
        #pragma unroll
        for (int nt = 0; nt < 4; ++nt)
            Ps[tt * 72 + nt * 16 + l15] = f2bf(sacc[nt][j] * r);
    }
    __syncthreads();

    // O = P V : K=64 (2 steps), N=32 (2 tiles)
    v8f oz = {0.f, 0.f, 0.f, 0.f, 0.f, 0.f, 0.f, 0.f};
    v8f oacc[2] = {oz, oz};
    const size_t vbase = qbase + 192;
    #pragma unroll
    for (int kstep = 0; kstep < 2; ++kstep) {
        FragBF pa;
        int row = mtile * 16 + l15;
        #pragma unroll
        for (int j = 0; j < 8; ++j)
            pa.u[j] = *(const unsigned int*)(&Ps[row * 72 + kstep * 32 + kAoff(j, half)]);
        #pragma unroll
        for (int vt = 0; vt < 2; ++vt) {
            FragBF vb;
            int d = vt * 16 + l15;
            #pragma unroll
            for (int j = 0; j < 8; ++j) {
                int s = kstep * 32 + kAoff(j, half);
                unsigned int lo = qkv[vbase + (size_t)s * 288 + d];
                unsigned int hi = qkv[vbase + (size_t)(s + 1) * 288 + d];
                vb.u[j] = lo | (hi << 16);
            }
            oacc[vt] = __builtin_amdgcn_wmma_f32_16x16x32_bf16(
                           false, pa.v, false, vb.v, (short)0, oacc[vt], false, false);
        }
    }
    #pragma unroll
    for (int vt = 0; vt < 2; ++vt) {
        #pragma unroll
        for (int j = 0; j < 8; ++j) {
            int tt = mtile * 16 + j + 8 * half;
            if (tt < NN) {
                int d = vt * 16 + l15;
                attn_out[((size_t)win * NN + tt) * CC + head * HD + d] = f2bf(oacc[vt][j]);
            }
        }
    }
}

// ---------------------------------------------------------------------------
// Host orchestration
// ---------------------------------------------------------------------------
extern "C" void kernel_launch(void* const* d_in, const int* in_sizes, int n_in,
                              void* d_out, int out_size, void* d_ws, size_t ws_size,
                              hipStream_t stream) {
    const float* x        = (const float*)d_in[0];
    const float* amask    = (const float*)d_in[1];
    const int*   relidx   = (const int*)  d_in[2];
    const float* n1g      = (const float*)d_in[3];
    const float* n1b      = (const float*)d_in[4];
    const float* qkv_w    = (const float*)d_in[5];
    const float* qkv_b    = (const float*)d_in[6];
    const float* proj_w   = (const float*)d_in[7];
    const float* proj_b   = (const float*)d_in[8];
    const float* rbt      = (const float*)d_in[9];
    const float* n2g      = (const float*)d_in[10];
    const float* n2b      = (const float*)d_in[11];
    const float* fc1_w    = (const float*)d_in[12];
    const float* fc1_b    = (const float*)d_in[13];
    const float* fc2_w    = (const float*)d_in[14];
    const float* fc2_b    = (const float*)d_in[15];
    float* out = (float*)d_out;
    char*  ws  = (char*)d_ws;

    // workspace layout (bytes)
    unsigned short* wts    = (unsigned short*)ws;                 // 110592 bf16
    unsigned short* qkvw_b = wts;
    unsigned short* projw_b= wts + 27648;
    unsigned short* fc1w_b = wts + 36864;
    unsigned short* fc2w_b = wts + 73728;
    unsigned short* xw     = (unsigned short*)(ws + 221184);                       // 401408*96
    unsigned short* qkvbuf = (unsigned short*)(ws + 221184 + 77070336ull);         // 401408*288
    unsigned short* attnb  = (unsigned short*)(ws + 221184 + 77070336ull + 231211008ull); // 401408*96
    unsigned short* fc1o   = qkvbuf;  // reuse: qkv+attn regions dead after proj
    unsigned short* h2     = xw;      // reuse: xw dead after QKV GEMM

    // 1) weights -> bf16
    conv_weights<<<432, 256, 0, stream>>>(qkv_w, proj_w, fc1_w, fc2_w, wts);
    // 2) LN1 + shift + window partition
    ln1_window<<<MTOT / 8, 256, 0, stream>>>(x, n1g, n1b, xw);
    // 3) QKV GEMM: (401408,96) x (288,96)^T -> bf16
    gemm_bf16<GM_BF16><<<(MTOT / 128) * (288 / 32), 256, 0, stream>>>(
        xw, qkvw_b, qkv_b, qkvbuf, nullptr, nullptr, MTOT, 288, CC);
    // 4) windowed attention (8192 windows x 3 heads)
    attn_kernel<<<NWIN * NHD, 128, 0, stream>>>(qkvbuf, relidx, rbt, amask, attnb);
    // 5) proj GEMM + window-reverse + shortcut -> d_out = x1
    gemm_bf16<GM_PROJ><<<(MTOT / 128) * (CC / 32), 256, 0, stream>>>(
        attnb, projw_b, proj_b, nullptr, out, x, MTOT, CC, CC);
    // 6) LN2 on x1
    ln2_kernel<<<MTOT / 8, 256, 0, stream>>>(out, n2g, n2b, h2);
    // 7) FC1 + GELU -> bf16
    gemm_bf16<GM_GELU><<<(MTOT / 128) * (HID / 32), 256, 0, stream>>>(
        h2, fc1w_b, fc1_b, fc1o, nullptr, nullptr, MTOT, HID, CC);
    // 8) FC2, accumulate into d_out (x1 + y)
    gemm_bf16<GM_FC2><<<(MTOT / 128) * (CC / 32), 256, 0, stream>>>(
        fc1o, fc2w_b, fc2_b, nullptr, out, nullptr, MTOT, CC, HID);
    (void)in_sizes; (void)n_in; (void)out_size; (void)ws_size;
}
